// MXMNet_63350767616124
// MI455X (gfx1250) — compile-verified, matrix operand order checked by reading, not run on hardware
//
#include <hip/hip_runtime.h>
#include <hip/hip_fp16.h>
#include <cmath>
#include <cstdint>
#include <cstddef>

// ---------------------------------------------------------------------------
// MXMNet forward for MI455X (gfx1250, wave32, WMMA).
// All dense layers run through a fused WMMA GEMM (v_wmma_f32_16x16x32_f16,
// f16 inputs / f32 accumulate) with epilogue fusion: +bias, SiLU, elementwise
// multiply, residual add.  Scatter/segment-sum via global_atomic_add_f32.
// The full weight matrix is staged once per block into dynamic LDS in the
// ISA B-fragment swizzle, so the K loop runs barrier-free with two
// ds_load_b128 per fragment.
// ---------------------------------------------------------------------------

#define NSPH 7
#define NRAD 6
#define NSBF 42
#define CUTOFF_F 5.0f

typedef __attribute__((ext_vector_type(16))) _Float16 v16h;
typedef __attribute__((ext_vector_type(8)))  float    v8f;
typedef __attribute__((ext_vector_type(4)))  float    v4f;

__device__ __forceinline__ float silu_f(float v) { return v / (1.f + __expf(-v)); }

// DimeNet envelope, p=6:  1/x - 28 x^5 + 48 x^6 - 21 x^7   (x < 1 else 0)
__device__ __forceinline__ float envelope6(float x) {
  if (x >= 1.f) return 0.f;
  float x2 = x * x;
  float x5 = x2 * x2 * x;
  return 1.f / x - 28.f * x5 + 48.f * x5 * x - 21.f * x5 * x2;
}

// ---------------------------------------------------------------------------
// Fused WMMA GEMM:  Y[N,128] = epi( X[N,K] @ W[K,128] )
//   MODE 0: X dense row-major [N,K]
//   MODE 1: X row r = concat( H[ia[r]][0:128], H[ib[r]][0:128], Xe[r][0:128] ),
//           K must be 384 (fused gather-concat for the edge MLPs)
//   epi(v) = ((silu?(v + bias[col])) * mul?[r,c]) + add?[r,c]
//
// Block = 128 threads = 4 wave32 waves; each wave computes a 16x128 strip
// (8 WMMA 16x16 tiles, 64 accumulator VGPRs).  The whole W matrix is staged
// once into dynamic LDS, pre-swizzled into the ISA B-fragment layout
// (CDNA5 ISA 7.12.2): per 32xK-block,
//   lane = (n&15) | (((k&15)>>3)<<4),  elem = (k&7) | ((k&16)>>1)
// so each lane fetches its v16h B fragment with two ds_read_b128.
// Dynamic LDS bytes at launch: ceil(K/32) * 8192  (max 96KB for K=384).
// ---------------------------------------------------------------------------
extern __shared__ _Float16 ldsW[];

template <int MODE>
__global__ __launch_bounds__(128) void gemm_wmma(
    const float* __restrict__ X, const int* __restrict__ ia,
    const int* __restrict__ ib, const float* __restrict__ Xe,
    const float* __restrict__ W, const float* __restrict__ bias,
    const float* __restrict__ mulp, const float* __restrict__ addp,
    float* __restrict__ Y, int N, int K, int act_silu) {
  const int tid = threadIdx.x;
  const int wave = tid >> 5;
  const int lane = tid & 31;
  const int rowbase = blockIdx.x * 64 + wave * 16;
  const int mrow = rowbase + (lane & 15);
  const int half = lane >> 4;
  const bool inrow = (mrow < N);

  // ---- stage ALL of W into LDS, swizzled into B-fragment layout -----------
  const int nkb = (K + 31) >> 5;
  for (int idx = tid; idx < nkb * 32 * 128; idx += 128) {
    int kb = idx >> 12;        // 32x128 block index
    int rem = idx & 4095;
    int kk = rem >> 7;         // 0..31
    int n = rem & 127;         // 0..127
    int kg = (kb << 5) + kk;
    float w = (kg < K) ? W[(long)kg * 128 + n] : 0.f;
    int tI = n >> 4;
    int lI = (n & 15) | (((kk & 15) >> 3) << 4);
    int eI = (kk & 7) | ((kk & 16) >> 1);
    ldsW[kb * 4096 + tI * 512 + lI * 16 + eI] = (_Float16)w;
  }

  const float* rowA = nullptr;
  const float* rowB = nullptr;
  const float* rowE = nullptr;
  {
    int r = inrow ? mrow : 0;
    if (MODE == 1) {
      rowA = X + (long)ia[r] * 128;
      rowB = X + (long)ib[r] * 128;
      rowE = Xe + (long)r * 128;
    } else {
      rowA = X + (long)r * K;
    }
  }

  v8f c[8];
#pragma unroll
  for (int t = 0; t < 8; ++t)
#pragma unroll
    for (int r = 0; r < 8; ++r) c[t][r] = 0.f;

  __syncthreads();
  const _Float16* ldsLane = ldsW + lane * 16;

  for (int k0 = 0; k0 < K; k0 += 32) {
    // --- build A fragment (16x32 f16): lane<16 -> K{0..7,16..23},
    //     lane>=16 -> K{8..15,24..31}, M = lane&15 --------------------------
    float fa[16];
    if (MODE == 0 && inrow && ((K & 3) == 0) && (k0 + 32 <= K)) {
      const float* base = rowA + k0 + half * 8;
      v4f u0 = *(const v4f*)(base);
      v4f u1 = *(const v4f*)(base + 4);
      v4f u2 = *(const v4f*)(base + 16);
      v4f u3 = *(const v4f*)(base + 20);
#pragma unroll
      for (int j = 0; j < 4; ++j) {
        fa[j] = u0[j]; fa[4 + j] = u1[j];
        fa[8 + j] = u2[j]; fa[12 + j] = u3[j];
      }
    } else {
#pragma unroll
      for (int j = 0; j < 8; ++j) {
        int kA0 = k0 + half * 8 + j;
        int kA1 = kA0 + 16;
        float f0 = 0.f, f1 = 0.f;
        if (inrow) {
          if (MODE == 0) {
            if (kA0 < K) f0 = rowA[kA0];
            if (kA1 < K) f1 = rowA[kA1];
          } else {
            f0 = (kA0 < 128) ? rowA[kA0]
                             : (kA0 < 256 ? rowB[kA0 - 128] : rowE[kA0 - 256]);
            f1 = (kA1 < 128) ? rowA[kA1]
                             : (kA1 < 256 ? rowB[kA1 - 128] : rowE[kA1 - 256]);
          }
        }
        fa[j] = f0;
        fa[8 + j] = f1;
      }
    }
    v16h a;
#pragma unroll
    for (int j = 0; j < 16; ++j) a[j] = (_Float16)fa[j];

    // --- 8 WMMA ops over the 128-wide output strip -------------------------
    const _Float16* wblk = ldsLane + (k0 >> 5) * 4096;
#pragma unroll
    for (int t = 0; t < 8; ++t) {
      v16h b = *(const v16h*)&wblk[t * 512];
      c[t] = __builtin_amdgcn_wmma_f32_16x16x32_f16(
          false, a, false, b, (short)0, c[t], false, false);
    }
  }

  // --- epilogue: C/D layout = lane&15 -> N col, VGPR r -> M row r (+8 hi) ---
  const int ncol0 = lane & 15;
  const int mofs = (lane >> 4) << 3;
  float bcol[8];
#pragma unroll
  for (int t = 0; t < 8; ++t) bcol[t] = bias ? bias[t * 16 + ncol0] : 0.f;
#pragma unroll
  for (int t = 0; t < 8; ++t) {
#pragma unroll
    for (int r = 0; r < 8; ++r) {
      int row = rowbase + mofs + r;
      if (row < N) {
        float v = c[t][r] + bcol[t];
        if (act_silu) v = silu_f(v);
        long off = (long)row * 128 + t * 16 + ncol0;
        if (mulp) v *= mulp[off];
        if (addp) v += addp[off];
        Y[off] = v;
      }
    }
  }
}

// ---------------------------------------------------------------------------
// element-wise / geometry / scatter kernels
// ---------------------------------------------------------------------------
struct SphRadC { float z[NSBF]; float nrm[NSBF]; };
struct SphPrefC { float p[NSPH]; };

__global__ void k_embed(const float* __restrict__ emb, const int* __restrict__ nt,
                        float* __restrict__ h, long n128) {
  long i = (long)blockIdx.x * blockDim.x + threadIdx.x;
  if (i < n128) h[i] = emb[(long)nt[i >> 7] * 128 + (i & 127)];
}

__global__ void k_dist(const float* __restrict__ pos, const int* __restrict__ jr,
                       const int* __restrict__ ir, float* __restrict__ d, long E) {
  long e = (long)blockIdx.x * blockDim.x + threadIdx.x;
  if (e >= E) return;
  int j = jr[e], i = ir[e];
  float dx = pos[(long)i * 3 + 0] - pos[(long)j * 3 + 0];
  float dy = pos[(long)i * 3 + 1] - pos[(long)j * 3 + 1];
  float dz = pos[(long)i * 3 + 2] - pos[(long)j * 3 + 2];
  d[e] = sqrtf(dx * dx + dy * dy + dz * dz);
}

__global__ void k_rbf16(const float* __restrict__ dist, const float* __restrict__ freq,
                        float* __restrict__ out, long E, float invcut) {
  long e = (long)blockIdx.x * blockDim.x + threadIdx.x;
  if (e >= E) return;
  float d = dist[e] * invcut;
  float env = envelope6(d);
#pragma unroll
  for (int k = 0; k < 16; ++k) out[e * 16 + k] = env * sinf(freq[k] * d);
}

__global__ void k_sbfrad(const float* __restrict__ dist, float* __restrict__ out,
                         long E, SphRadC C) {
  long e = (long)blockIdx.x * blockDim.x + threadIdx.x;
  if (e >= E) return;
  float d = dist[e] * (1.f / CUTOFF_F);
  float env = envelope6(d);
#pragma unroll
  for (int n = 0; n < NSPH; ++n) {
#pragma unroll
    for (int k = 0; k < NRAD; ++k) {
      float x = d * C.z[n * NRAD + k];
      float s = sinf(x), cx = cosf(x);
      float j0 = s / x;
      float j;
      if (n == 0) {
        j = j0;
      } else {
        float j1 = s / (x * x) - cx / x;
        float jm = j0, jc = j1;
        for (int m = 1; m < n; ++m) {
          float t = (2 * m + 1) / x * jc - jm;
          jm = jc; jc = t;
        }
        j = jc;
      }
      out[e * NSBF + n * NRAD + k] = C.nrm[n * NRAD + k] * j * env;
    }
  }
}

__global__ void k_angle(const float* __restrict__ pos, const int* __restrict__ ia,
                        const int* __restrict__ ib, const int* __restrict__ ic,
                        float* __restrict__ ang, long T) {
  long t = (long)blockIdx.x * blockDim.x + threadIdx.x;
  if (t >= T) return;
  long a = ia[t], b = ib[t], c = ic[t];
  float v1x = pos[b * 3 + 0] - pos[a * 3 + 0];
  float v1y = pos[b * 3 + 1] - pos[a * 3 + 1];
  float v1z = pos[b * 3 + 2] - pos[a * 3 + 2];
  float v2x = pos[c * 3 + 0] - pos[b * 3 + 0];
  float v2y = pos[c * 3 + 1] - pos[b * 3 + 1];
  float v2z = pos[c * 3 + 2] - pos[b * 3 + 2];
  float cx = v1y * v2z - v1z * v2y;
  float cy = v1z * v2x - v1x * v2z;
  float cz = v1x * v2y - v1y * v2x;
  float cr = sqrtf(cx * cx + cy * cy + cz * cz);
  float dt = v1x * v2x + v1y * v2y + v1z * v2z;
  ang[t] = atan2f(cr, dt);
}

__global__ void k_sbfcomb(const float* __restrict__ rad, const int* __restrict__ idxe,
                          const float* __restrict__ ang, float* __restrict__ out,
                          long T, SphPrefC C) {
  long t = (long)blockIdx.x * blockDim.x + threadIdx.x;
  if (t >= T) return;
  float cth = cosf(ang[t]);
  float P[NSPH];
  P[0] = 1.f;
  P[1] = cth;
#pragma unroll
  for (int n = 1; n < NSPH - 1; ++n)
    P[n + 1] = ((2 * n + 1) * cth * P[n] - n * P[n - 1]) / (float)(n + 1);
  const float* r = rad + (long)idxe[t] * NSBF;
#pragma unroll
  for (int n = 0; n < NSPH; ++n) {
    float cb = C.p[n] * P[n];
#pragma unroll
    for (int k = 0; k < NRAD; ++k)
      out[t * NSBF + n * NRAD + k] = r[n * NRAD + k] * cb;
  }
}

__global__ void k_copy(const float* __restrict__ s, float* __restrict__ d, long n) {
  long i = (long)blockIdx.x * blockDim.x + threadIdx.x;
  if (i < n) d[i] = s[i];
}

__global__ void k_zero(float* __restrict__ d, long n) {
  long i = (long)blockIdx.x * blockDim.x + threadIdx.x;
  if (i < n) d[i] = 0.f;
}

// dst[seg[e], c] += src[e, c]   (hardware global_atomic_add_f32)
__global__ void k_scatter_cols(const float* __restrict__ src, const int* __restrict__ seg,
                               float* __restrict__ dst, long E) {
  long i = (long)blockIdx.x * blockDim.x + threadIdx.x;
  if (i >= E * 128) return;
  long e = i >> 7;
  int c = i & 127;
  unsafeAtomicAdd(&dst[(long)seg[e] * 128 + c], src[i]);
}

// dst[sidx[t], c] += msrc[gidx[t], c] * s[t, c]
__global__ void k_gms(const float* __restrict__ msrc, const int* __restrict__ gidx,
                      const float* __restrict__ s, const int* __restrict__ sidx,
                      float* __restrict__ dst, long T) {
  long i = (long)blockIdx.x * blockDim.x + threadIdx.x;
  if (i >= T * 128) return;
  long t = i >> 7;
  int c = i & 127;
  float v = msrc[(long)gidx[t] * 128 + c] * s[i];
  unsafeAtomicAdd(&dst[(long)sidx[t] * 128 + c], v);
}

// nsum[n] += y[n,:] . w + b
__global__ void k_yred(const float* __restrict__ y, const float* __restrict__ w,
                       const float* __restrict__ b, float* __restrict__ nsum, int N) {
  int n = blockIdx.x * blockDim.x + threadIdx.x;
  if (n >= N) return;
  float s = b[0];
  const float* row = y + (long)n * 128;
#pragma unroll 8
  for (int c = 0; c < 128; ++c) s += row[c] * w[c];
  nsum[n] += s;
}

__global__ void k_graphsum(const float* __restrict__ ns, float* __restrict__ out,
                           int nG, int atoms) {
  int g = blockIdx.x * blockDim.x + threadIdx.x;
  if (g >= nG) return;
  float s = 0.f;
  for (int a = 0; a < atoms; ++a) s += ns[(long)g * atoms + a];
  out[g] = s;
}

// ---------------------------------------------------------------------------
// host: spherical-Bessel roots/norms (double-precision bisection, per call,
// deterministic) passed to device by value.
// ---------------------------------------------------------------------------
static double jn_host(double x, int n) {
  double j0 = sin(x) / x;
  if (n == 0) return j0;
  double j1 = sin(x) / (x * x) - cos(x) / x;
  double jm = j0, jc = j1;
  for (int m = 1; m < n; ++m) {
    double t = (2 * m + 1) / x * jc - jm;
    jm = jc; jc = t;
  }
  return jc;
}

static void build_sph(SphRadC* rc, SphPrefC* pc) {
  const double PI = 3.14159265358979323846;
  double zeros[NSPH][NSPH + NRAD];
  int cnt[NSPH];
  for (int i = 0; i < NSPH + NRAD; ++i) zeros[0][i] = PI * (i + 1);
  cnt[0] = NSPH + NRAD;
  for (int order = 1; order < NSPH; ++order) {
    int pcn = cnt[order - 1];
    for (int i = 0; i < pcn - 1; ++i) {
      double a = zeros[order - 1][i], b = zeros[order - 1][i + 1];
      double fa = jn_host(a, order);
      for (int it = 0; it < 100; ++it) {
        double m = 0.5 * (a + b), fm = jn_host(m, order);
        if ((fm > 0) == (fa > 0)) { a = m; fa = fm; } else { b = m; }
      }
      zeros[order][i] = 0.5 * (a + b);
    }
    cnt[order] = pcn - 1;
  }
  for (int n = 0; n < NSPH; ++n) {
    for (int k = 0; k < NRAD; ++k) {
      double z = zeros[n][k];
      rc->z[n * NRAD + k] = (float)z;
      rc->nrm[n * NRAD + k] = (float)(sqrt(2.0) / fabs(jn_host(z, n + 1)));
    }
    pc->p[n] = (float)sqrt((2.0 * n + 1) / (4.0 * PI));
  }
}

// ---------------------------------------------------------------------------
static inline int cdiv(long a, long b) { return (int)((a + b - 1) / b); }

static void gemm(hipStream_t st, const float* X, const float* W, const float* bias,
                 const float* mul, const float* add, float* Y, long N, int K, int act) {
  if (N <= 0) return;
  size_t sh = (size_t)((K + 31) / 32) * 8192;
  gemm_wmma<0><<<dim3(cdiv(N, 64)), dim3(128), sh, st>>>(
      X, nullptr, nullptr, nullptr, W, bias, mul, add, Y, (int)N, K, act);
}

static void gemm_cat(hipStream_t st, const float* H, const int* ia, const int* ib,
                     const float* Xe, const float* W, const float* bias,
                     const float* mul, const float* add, float* Y, long N, int act) {
  if (N <= 0) return;
  size_t sh = (size_t)(384 / 32) * 8192;  // 96KB
  gemm_wmma<1><<<dim3(cdiv(N, 64)), dim3(128), sh, st>>>(
      H, ia, ib, Xe, W, bias, mul, add, Y, (int)N, 384, act);
}

extern "C" void kernel_launch(void* const* d_in, const int* in_sizes, int n_in,
                              void* d_out, int out_size, void* d_ws, size_t ws_size,
                              hipStream_t stream) {
  // --------- input flatten assumption -------------------------------------
  // Depth-first, dict-insertion order of setup_inputs():
  //   params leaves 0..190 (emb, freq_g, freq_l, rbf_g_mlp{W,b}, rbf_l_mlp{W,b},
  //     sbf1_mlp{W,b}, sbf2_mlp{W,b}, global[0..2] x19, local[0..2] x41),
  //   pos=191, node_type=192, edge_index_l=193, edge_index_g=194,
  //   idx_i_1=195, idx_j=196, idx_k=197, idx_kj=198, idx_ji_1=199,
  //   idx_i_2=200, idx_j1=201, idx_j2=202, idx_jj=203, idx_ji_2=204, batch=205
  auto F = [&](int i) { return (const float*)d_in[i]; };
  auto I = [&](int i) { return (const int*)d_in[i]; };

  const float* emb = F(0);
  const float* freq_g = F(1);
  const float* freq_l = F(2);
  const float* rbfgW = F(3);  const float* rbfgB = F(4);
  const float* rbflW = F(5);  const float* rbflB = F(6);
  const float* sbf1W = F(7);  const float* sbf1B = F(8);
  const float* sbf2W = F(9);  const float* sbf2B = F(10);
  const int PG = 11;             // global layers: 19 leaves each
  const int PL = PG + 3 * 19;    // local layers:  41 leaves each
  auto G  = [&](int L, int o) { return F(PG + L * 19 + o); };
  auto Lp = [&](int L, int o) { return F(PL + L * 41 + o); };
  // global offs: hW=0 hb=1 | res1:2-5 res2:6-9 res3:10-13 | mlpW=14 mlpb=15
  //              xeW=16 xeb=17 | lin=18
  // local offs:  hW=0 hb=1 | kjW=2 kjb=3 | ji1W=4 ji1b=5 | ji2W=6 ji2b=7
  //              jjW=8 jjb=9 | sbf1:10-13 sbf2:14-17 | linr1=18 linr2=19
  //              linout=20 | res1:21-24 res2:25-28 res3:29-32
  //              ymlp:33-38 | yW=39 yb=40

  const float* pos = F(191);
  const int* ntype = I(192);
  const int N = in_sizes[192];
  const long El = in_sizes[193] / 2;
  const long Eg = in_sizes[194] / 2;
  const int* jl = I(193);           // edge_index_l row 0
  const int* il = I(193) + El;      // edge_index_l row 1
  const int* jg = I(194);
  const int* ig = I(194) + Eg;
  const int* idx_i1 = I(195);
  const int* idx_j  = I(196);
  const int* idx_k  = I(197);
  const int* idx_kj = I(198);  const long T1 = in_sizes[198];
  const int* idx_ji1 = I(199);
  const int* idx_i2 = I(200);
  const int* idx_j1 = I(201);
  const int* idx_j2 = I(202);
  const int* idx_jj = I(203);  const long T2 = in_sizes[203];
  const int* idx_ji2 = I(204);
  const int nG = out_size;
  const int atoms = (nG > 0) ? (N / nG) : 1;
  const long Tm = (T1 > T2) ? T1 : T2;

  // --------- workspace bump allocator -------------------------------------
  char* wp = (char*)d_ws;
  auto alloc = [&](size_t elems) -> float* {
    float* r = (float*)wp;
    wp += ((elems * sizeof(float)) + 255) & ~(size_t)255;
    return r;
  };
  float* h    = alloc((size_t)N * 128);
  float* hA   = alloc((size_t)N * 128);
  float* hB   = alloc((size_t)N * 128);
  float* hC   = alloc((size_t)N * 128);
  float* hD   = alloc((size_t)N * 128);
  float* nsum = alloc((size_t)N);
  float* dg   = alloc((size_t)Eg);
  float* dl   = alloc((size_t)El);
  float* rbfg = alloc((size_t)Eg * 128);
  float* rbfl = alloc((size_t)El * 128);
  float* eA   = alloc((size_t)Eg * 128);   // also reused as rbf16 raw (E_g x16)
  float* eB   = alloc((size_t)Eg * 128);
  float* srad = alloc((size_t)El * NSBF);
  float* an1  = alloc((size_t)T1);
  float* an2  = alloc((size_t)T2);
  float* s1r  = alloc((size_t)T1 * NSBF);
  float* s2r  = alloc((size_t)T2 * NSBF);
  float* sbf1 = alloc((size_t)T1 * 128);
  float* sbf2 = alloc((size_t)T2 * 128);
  float* tA   = alloc((size_t)Tm * 128);
  float* tB   = alloc((size_t)Tm * 128);
  float* tE   = alloc((size_t)El * 128);   // also reused as rbf16 raw (E_l x16)
  float* m1   = alloc((size_t)El * 128);
  float* m2   = alloc((size_t)El * 128);
  float* m3   = alloc((size_t)El * 128);
  (void)ws_size; (void)n_in;

  SphRadC sph; SphPrefC pref;
  build_sph(&sph, &pref);

  const int B = 256;
  auto gl = [&](long n) { return dim3(cdiv(n, B)); };

  // --------- geometry + basis ---------------------------------------------
  k_embed<<<gl((long)N * 128), B, 0, stream>>>(emb, ntype, h, (long)N * 128);
  if (Eg > 0) k_dist<<<gl(Eg), B, 0, stream>>>(pos, jg, ig, dg, Eg);
  if (El > 0) k_dist<<<gl(El), B, 0, stream>>>(pos, jl, il, dl, El);
  if (Eg > 0) k_rbf16<<<gl(Eg), B, 0, stream>>>(dg, freq_g, eA, Eg, 1.f / CUTOFF_F);
  gemm(stream, eA, rbfgW, rbfgB, nullptr, nullptr, rbfg, Eg, 16, 1);
  if (El > 0) k_rbf16<<<gl(El), B, 0, stream>>>(dl, freq_l, tE, El, 1.f / 5.0f);
  gemm(stream, tE, rbflW, rbflB, nullptr, nullptr, rbfl, El, 16, 1);
  if (El > 0) k_sbfrad<<<gl(El), B, 0, stream>>>(dl, srad, El, sph);
  if (T1 > 0) k_angle<<<gl(T1), B, 0, stream>>>(pos, idx_i1, idx_j, idx_k, an1, T1);
  if (T2 > 0) k_angle<<<gl(T2), B, 0, stream>>>(pos, idx_i2, idx_j1, idx_j2, an2, T2);
  if (T1 > 0) k_sbfcomb<<<gl(T1), B, 0, stream>>>(srad, idx_kj, an1, s1r, T1, pref);
  if (T2 > 0) k_sbfcomb<<<gl(T2), B, 0, stream>>>(srad, idx_jj, an2, s2r, T2, pref);
  gemm(stream, s1r, sbf1W, sbf1B, nullptr, nullptr, sbf1, T1, NSBF, 1);
  gemm(stream, s2r, sbf2W, sbf2B, nullptr, nullptr, sbf2, T2, NSBF, 1);
  k_zero<<<gl(N), B, 0, stream>>>(nsum, N);

  // --------- 3 message-passing layers -------------------------------------
  for (int L = 0; L < 3; ++L) {
    // ===== global MP ===== (res_h lives in h)
    gemm(stream, h, G(L, 0), G(L, 1), nullptr, nullptr, hA, N, 128, 1);   // h_mlp
    // prop 1: xe = silu([h[ig],h[jg],rbf] @ Wx + bx); msg = (rbf@lin) * xe
    gemm_cat(stream, hA, ig, jg, rbfg, G(L, 16), G(L, 17), nullptr, nullptr, eA, Eg, 1);
    gemm(stream, rbfg, G(L, 18), nullptr, eA, nullptr, eB, Eg, 128, 0);
    k_copy<<<gl((long)N * 128), B, 0, stream>>>(hA, hB, (long)N * 128);
    if (Eg > 0) k_scatter_cols<<<gl(Eg * 128), B, 0, stream>>>(eB, ig, hB, Eg);
    // res1
    gemm(stream, hB, G(L, 2), G(L, 3), nullptr, nullptr, hC, N, 128, 1);
    gemm(stream, hC, G(L, 4), G(L, 5), nullptr, hB, hB, N, 128, 1);
    // h = mlp(h) + res_h
    gemm(stream, hB, G(L, 14), G(L, 15), nullptr, h, hA, N, 128, 1);
    // res2, res3
    gemm(stream, hA, G(L, 6), G(L, 7), nullptr, nullptr, hC, N, 128, 1);
    gemm(stream, hC, G(L, 8), G(L, 9), nullptr, hA, hA, N, 128, 1);
    gemm(stream, hA, G(L, 10), G(L, 11), nullptr, nullptr, hC, N, 128, 1);
    gemm(stream, hC, G(L, 12), G(L, 13), nullptr, hA, hA, N, 128, 1);
    // prop 2 -> h
    gemm_cat(stream, hA, ig, jg, rbfg, G(L, 16), G(L, 17), nullptr, nullptr, eA, Eg, 1);
    gemm(stream, rbfg, G(L, 18), nullptr, eA, nullptr, eB, Eg, 128, 0);
    k_copy<<<gl((long)N * 128), B, 0, stream>>>(hA, h, (long)N * 128);
    if (Eg > 0) k_scatter_cols<<<gl(Eg * 128), B, 0, stream>>>(eB, ig, h, Eg);

    // ===== local MP ===== (res_h lives in h)
    gemm(stream, h, Lp(L, 0), Lp(L, 1), nullptr, nullptr, hA, N, 128, 1);  // h_mlp
    // m_kj = silu([h[il],h[jl],rbf]@Wkj+b) * (rbf@lin_rbf1)
    gemm(stream, rbfl, Lp(L, 18), nullptr, nullptr, nullptr, tE, El, 128, 0);
    gemm_cat(stream, hA, il, jl, rbfl, Lp(L, 2), Lp(L, 3), tE, nullptr, m1, El, 1);
    // s1 = mlp_sbf1(sbf1)
    gemm(stream, sbf1, Lp(L, 10), Lp(L, 11), nullptr, nullptr, tA, T1, 128, 1);
    gemm(stream, tA, Lp(L, 12), Lp(L, 13), nullptr, nullptr, tB, T1, 128, 1);
    // m = mlp_ji_1(m) + segsum(m_kj[idx_kj]*s1, idx_ji_1)
    gemm_cat(stream, hA, il, jl, rbfl, Lp(L, 4), Lp(L, 5), nullptr, nullptr, m2, El, 1);
    if (T1 > 0) k_gms<<<gl(T1 * 128), B, 0, stream>>>(m1, idx_kj, tB, idx_ji1, m2, T1);
    // m_jj = mlp_jj(m) * (rbf@lin_rbf2)
    gemm(stream, rbfl, Lp(L, 19), nullptr, nullptr, nullptr, tE, El, 128, 0);
    gemm(stream, m2, Lp(L, 8), Lp(L, 9), tE, nullptr, m1, El, 128, 1);
    // s2 = mlp_sbf2(sbf2)
    gemm(stream, sbf2, Lp(L, 14), Lp(L, 15), nullptr, nullptr, tA, T2, 128, 1);
    gemm(stream, tA, Lp(L, 16), Lp(L, 17), nullptr, nullptr, tB, T2, 128, 1);
    // m = mlp_ji_2(m) + segsum(m_jj[idx_jj]*s2, idx_ji_2)
    gemm(stream, m2, Lp(L, 6), Lp(L, 7), nullptr, nullptr, m3, El, 128, 1);
    if (T2 > 0) k_gms<<<gl(T2 * 128), B, 0, stream>>>(m1, idx_jj, tB, idx_ji2, m3, T2);
    // m = (rbf@lin_rbf_out) * m ; h = segsum(m, il)
    gemm(stream, rbfl, Lp(L, 20), nullptr, m3, nullptr, m2, El, 128, 0);
    k_zero<<<gl((long)N * 128), B, 0, stream>>>(hB, (long)N * 128);
    if (El > 0) k_scatter_cols<<<gl(El * 128), B, 0, stream>>>(m2, il, hB, El);
    // res1 ; h = h_mlp(h) + res_h ; res2 ; res3
    gemm(stream, hB, Lp(L, 21), Lp(L, 22), nullptr, nullptr, hC, N, 128, 1);
    gemm(stream, hC, Lp(L, 23), Lp(L, 24), nullptr, hB, hB, N, 128, 1);
    gemm(stream, hB, Lp(L, 0), Lp(L, 1), nullptr, h, h, N, 128, 1);
    gemm(stream, h, Lp(L, 25), Lp(L, 26), nullptr, nullptr, hC, N, 128, 1);
    gemm(stream, hC, Lp(L, 27), Lp(L, 28), nullptr, h, h, N, 128, 1);
    gemm(stream, h, Lp(L, 29), Lp(L, 30), nullptr, nullptr, hC, N, 128, 1);
    gemm(stream, hC, Lp(L, 31), Lp(L, 32), nullptr, h, h, N, 128, 1);
    // y = y_mlp(h) @ yW + yb ; node_sum += y
    gemm(stream, h, Lp(L, 33), Lp(L, 34), nullptr, nullptr, hC, N, 128, 1);
    gemm(stream, hC, Lp(L, 35), Lp(L, 36), nullptr, nullptr, hD, N, 128, 1);
    gemm(stream, hD, Lp(L, 37), Lp(L, 38), nullptr, nullptr, hC, N, 128, 1);
    k_yred<<<gl(N), B, 0, stream>>>(hC, Lp(L, 39), Lp(L, 40), nsum, N);
  }

  // --------- per-graph reduction ------------------------------------------
  k_graphsum<<<gl(nG), B, 0, stream>>>(nsum, (float*)d_out, nG, atoms);
}
